// CrossRPEAttention_40862318854716
// MI455X (gfx1250) — matrix operand order repaired and checked by reading, not compiled, not run on hardware
//
#include <hip/hip_runtime.h>

typedef __attribute__((ext_vector_type(16))) __bf16 v16bf;
typedef __attribute__((ext_vector_type(8)))  __bf16 v8bf;
typedef __attribute__((ext_vector_type(8)))  float  v8f;
typedef __attribute__((ext_vector_type(4)))  int    v4i;

// Direct global->LDS async copy path (CDNA5, ASYNCcnt-tracked). Guarded so the
// host pass and any toolchain lacking the builtin fall back to VGPR-relay.
#if defined(__AMDGCN__) && \
    __has_builtin(__builtin_amdgcn_global_load_async_to_lds_b128) && \
    __has_builtin(__builtin_amdgcn_s_wait_asynccnt)
#define HAVE_ASYNC_LDS 1
#else
#define HAVE_ASYNC_LDS 0
#endif

#define AS_GLOBAL __attribute__((address_space(1)))
#define AS_LDS    __attribute__((address_space(3)))

namespace cfg {
constexpr int B    = 64;
constexpr int Hh   = 12;
constexpr int HD   = 64;
constexpr int Cc   = 768;
constexpr int Nn   = 393;   // 1 + 196*2
constexpr int Pp   = 197;
constexpr int PPAD = 208;   // 13 * 16
constexpr int NPAD = 416;   // 13 * 32
constexpr int PTILES = PPAD / 16;
constexpr int NTILES = NPAD / 16;
}

__device__ __forceinline__ v8f wmma_bf16f32(v16bf a, v16bf b, v8f c) {
  // D(f32 16x16) = A(bf16 16x32) * B(bf16 32x16) + C
  return __builtin_amdgcn_wmma_f32_16x16x32_bf16(false, a, false, b, (short)0, c,
                                                 false, false);
}

// A-matrix 16x32 bf16 per-lane layout: lane-half 0 holds K
// {0..15}, lane-half 1 holds K {16..31}. rowp = row (K-contig).
__device__ __forceinline__ v16bf load_a_tile(const __bf16* rowp, int kbase, int lhalf) {
  return *(const v16bf*)(rowp + kbase + lhalf * 16);
}
// B-matrix 32x16 bf16: lane = output column; K contiguous (lanes 0-15: K0..15,
// lanes 16-31: K16..31). colp = the column's K-contiguous vector in memory.
__device__ __forceinline__ v16bf load_b_tile(const __bf16* colp, int kbase, int lhalf) {
  return *(const v16bf*)(colp + kbase + lhalf * 16);
}

#if HAVE_ASYNC_LDS
__device__ __forceinline__ void async_copy_b128(const void* gsrc, void* ldsdst) {
  __builtin_amdgcn_global_load_async_to_lds_b128(
      (AS_GLOBAL v4i*)(AS_GLOBAL void*)gsrc,
      (AS_LDS v4i*)(AS_LDS void*)ldsdst, 0, 0);
}
#endif

__global__ void cvt_bf16_kernel(const float* __restrict__ in,
                                __bf16* __restrict__ out, int n) {
  int i = blockIdx.x * 256 + threadIdx.x;
  if (i < n) out[i] = (__bf16)in[i];
}

// Batch-invariant: qbf (H, PPAD, 64) bf16 and bias (H, PPAD, NPAD) f32.
__global__ __launch_bounds__(128) void prep_q_bias_kernel(
    const float* __restrict__ q_learned, const float* __restrict__ pos_embed,
    const float* __restrict__ rpe_w, const int* __restrict__ rp_bucket,
    __bf16* __restrict__ qbf, float* __restrict__ bias) {
  using namespace cfg;
  const int p = blockIdx.x;   // 0..PPAD-1
  const int h = blockIdx.y;   // 0..11
  const int tid = threadIdx.x;
  __shared__ float qs[HD];
  __shared__ float lk[8];     // n_buckets = 2*3+1+1 = 8
  if (tid < HD) {
    float v = 0.0f;
    if (p < Pp) v = q_learned[h * HD + tid] + pos_embed[(size_t)p * Cc + h * HD + tid];
    qs[tid] = v;
    qbf[((size_t)h * PPAD + p) * HD + tid] = (__bf16)v;
  }
  __syncthreads();
  if (tid < 8) {
    float s = 0.0f;
    for (int d = 0; d < HD; ++d) s += qs[d] * rpe_w[d * 8 + tid];
    lk[tid] = s;
  }
  __syncthreads();
  const int pp = p < Pp ? p : Pp - 1;
  const int* rb = rp_bucket + (size_t)pp * Pp;
  float* brow = bias + ((size_t)h * PPAD + p) * NPAD;
  for (int n = tid; n < NPAD; n += 128) {
    float bv;
    if (n >= Nn) bv = -1e30f;                      // softmax mask for pad cols
    else {
      int j = (n == 0) ? 0 : 1 + (n - 1) % (Pp - 1);  // tiled RPE columns
      bv = lk[rb[j]];
    }
    brow[n] = bv;
  }
}

// K = x@wk^T  -> (b,h,n,d) bf16 ; V = x@wv^T stored transposed -> (b,h,d,n) bf16.
__global__ __launch_bounds__(128) void kv_proj_kernel(
    const __bf16* __restrict__ xbf, const __bf16* __restrict__ wkbf,
    const __bf16* __restrict__ wvbf, __bf16* __restrict__ Kbf,
    __bf16* __restrict__ Vt) {
  using namespace cfg;
  const int ntile = blockIdx.x;   // 0..NTILES-1 over NPAD rows
  const int b = blockIdx.y;
  const int tid = threadIdx.x;
  __shared__ __bf16 As[16][Cc];
  for (int ch = tid; ch < 16 * (Cc / 8); ch += 128) {
    int row = ch / (Cc / 8);
    int co = (ch % (Cc / 8)) * 8;
    int n = ntile * 16 + row;
    if (n < Nn) {
#if HAVE_ASYNC_LDS
      async_copy_b128(xbf + ((size_t)b * Nn + n) * Cc + co, &As[row][co]);
#else
      *(uint4*)(&As[row][co]) =
          *(const uint4*)(xbf + ((size_t)b * Nn + n) * Cc + co);
#endif
    } else {
      *(uint4*)(&As[row][co]) = make_uint4(0u, 0u, 0u, 0u);  // pad rows -> zeros
    }
  }
#if HAVE_ASYNC_LDS
  __builtin_amdgcn_s_wait_asynccnt(0);
#endif
  __syncthreads();
  const int wave = tid >> 5, lane = tid & 31;
  const int lrow = lane & 15, lhalf = lane >> 4;
  for (int t = wave; t < 96; t += 4) {          // 48 K col-tiles + 48 V col-tiles
    const bool isK = (t < 48);
    const int ct = isK ? t : t - 48;
    const __bf16* W = (isK ? wkbf : wvbf) + (size_t)(ct * 16 + lrow) * Cc;
    v8f acc = {};
    for (int ks = 0; ks < Cc / 32; ++ks) {
      v16bf a = load_a_tile(&As[lrow][0], ks * 32, lhalf);
      v16bf w = load_b_tile(W, ks * 32, lhalf);
      acc = wmma_bf16f32(a, w, acc);
    }
    const int h = ct >> 2;
    const int d = (ct & 3) * 16 + lrow;
    for (int r = 0; r < 8; ++r) {
      int n = ntile * 16 + lhalf * 8 + r;
      __bf16 val = (__bf16)acc[r];
      if (isK) Kbf[(((size_t)b * Hh + h) * NPAD + n) * HD + d] = val;
      else     Vt[(((size_t)b * Hh + h) * HD + d) * NPAD + n] = val;
    }
  }
}

// Fused QK^T + bias + softmax + PV for one (b, h, 16-row q tile).
__global__ __launch_bounds__(128) void attn_kernel(
    const __bf16* __restrict__ qbf, const __bf16* __restrict__ Kbf,
    const __bf16* __restrict__ Vt, const float* __restrict__ bias,
    __bf16* __restrict__ Obf) {
  using namespace cfg;
  const int ptile = blockIdx.x;
  const int h = blockIdx.y;
  const int b = blockIdx.z;
  const int tid = threadIdx.x;
  const int wave = tid >> 5, lane = tid & 31;
  const int lrow = lane & 15, lhalf = lane >> 4;
  __shared__ float  S[16][NPAD];
  __shared__ __bf16 Pb[16][NPAD];

  const __bf16* qrow = qbf + ((size_t)h * PPAD + ptile * 16 + lrow) * HD;
  v16bf a0 = load_a_tile(qrow, 0, lhalf);
  v16bf a1 = load_a_tile(qrow, 32, lhalf);
  const __bf16* Kb = Kbf + ((size_t)b * Hh + h) * NPAD * HD;
  const float* brow = bias + ((size_t)h * PPAD + ptile * 16) * NPAD;
  constexpr float scale = 0.125f;   // 64^-0.5
  for (int nt = wave; nt < NTILES; nt += 4) {
    const __bf16* krow = Kb + (size_t)(nt * 16 + lrow) * HD;
    v16bf b0 = load_b_tile(krow, 0, lhalf);
    v16bf b1 = load_b_tile(krow, 32, lhalf);
    v8f acc = {};
    acc = wmma_bf16f32(a0, b0, acc);
    acc = wmma_bf16f32(a1, b1, acc);
    const int col = nt * 16 + lrow;
    for (int r = 0; r < 8; ++r) {
      int m = lhalf * 8 + r;
      S[m][col] = acc[r] * scale + brow[(size_t)m * NPAD + col];
    }
  }
  __syncthreads();
  {   // softmax: 8 lanes per row, width-8 xor reductions
    const int row = tid >> 3, k8 = tid & 7;
    float* Srow = &S[row][0];
    float mx = -1e30f;
    for (int c = k8; c < NPAD; c += 8) mx = fmaxf(mx, Srow[c]);
    for (int off = 4; off; off >>= 1) mx = fmaxf(mx, __shfl_xor(mx, off, 8));
    float sum = 0.0f;
    for (int c = k8; c < NPAD; c += 8) {
      float e = __expf(Srow[c] - mx);
      Srow[c] = e;
      sum += e;
    }
    for (int off = 4; off; off >>= 1) sum += __shfl_xor(sum, off, 8);
    float inv = 1.0f / sum;
    for (int c = k8; c < NPAD; c += 8) Pb[row][c] = (__bf16)(Srow[c] * inv);
  }
  __syncthreads();
  // PV: each wave owns one 16-wide d tile; contract over NPAD in steps of 32.
  const int dt = wave;
  const __bf16* Vrow = Vt + (((size_t)b * Hh + h) * HD + dt * 16 + lrow) * NPAD;
  v8f o = {};
  for (int ns = 0; ns < NPAD / 32; ++ns) {
    v16bf pa = load_a_tile(&Pb[lrow][0], ns * 32, lhalf);
    v16bf vb = load_b_tile(Vrow, ns * 32, lhalf);
    o = wmma_bf16f32(pa, vb, o);
  }
  __bf16* obase = Obf + ((size_t)b * PPAD + ptile * 16) * Cc + h * HD + dt * 16 + lrow;
  for (int r = 0; r < 8; ++r) obase[(size_t)(lhalf * 8 + r) * Cc] = (__bf16)o[r];
}

// out = Obf @ proj_w^T + proj_b  (f32 output, only valid p < 197 stored)
__global__ __launch_bounds__(128) void out_proj_kernel(
    const __bf16* __restrict__ Obf, const __bf16* __restrict__ pwbf,
    const float* __restrict__ proj_b, float* __restrict__ out) {
  using namespace cfg;
  const int ptile = blockIdx.x;
  const int b = blockIdx.y;
  const int tid = threadIdx.x;
  __shared__ __bf16 As[16][Cc];
  for (int ch = tid; ch < 16 * (Cc / 8); ch += 128) {
    int row = ch / (Cc / 8);
    int co = (ch % (Cc / 8)) * 8;
#if HAVE_ASYNC_LDS
    async_copy_b128(Obf + ((size_t)b * PPAD + ptile * 16 + row) * Cc + co,
                    &As[row][co]);
#else
    *(uint4*)(&As[row][co]) =
        *(const uint4*)(Obf + ((size_t)b * PPAD + ptile * 16 + row) * Cc + co);
#endif
  }
#if HAVE_ASYNC_LDS
  __builtin_amdgcn_s_wait_asynccnt(0);
#endif
  __syncthreads();
  const int wave = tid >> 5, lane = tid & 31;
  const int lrow = lane & 15, lhalf = lane >> 4;
  for (int t = wave; t < 48; t += 4) {
    const __bf16* W = pwbf + (size_t)(t * 16 + lrow) * Cc;
    v8f acc = {};
    for (int ks = 0; ks < Cc / 32; ++ks) {
      v16bf a = load_a_tile(&As[lrow][0], ks * 32, lhalf);
      v16bf w = load_b_tile(W, ks * 32, lhalf);
      acc = wmma_bf16f32(a, w, acc);
    }
    const int cp = t * 16 + lrow;
    const float bb = proj_b[cp];
    for (int r = 0; r < 8; ++r) {
      int p = ptile * 16 + lhalf * 8 + r;
      if (p < Pp) out[((size_t)b * Pp + p) * Cc + cp] = acc[r] + bb;
    }
  }
}

extern "C" void kernel_launch(void* const* d_in, const int* in_sizes, int n_in,
                              void* d_out, int out_size, void* d_ws, size_t ws_size,
                              hipStream_t stream) {
  using namespace cfg;
  const float* x         = (const float*)d_in[0];
  const float* q_learned = (const float*)d_in[1];
  const float* pos_embed = (const float*)d_in[2];
  const float* wk        = (const float*)d_in[3];
  const float* wv        = (const float*)d_in[4];
  const float* rpe_w     = (const float*)d_in[5];
  const float* proj_w    = (const float*)d_in[6];
  const float* proj_b    = (const float*)d_in[7];
  const int*   rp_bucket = (const int*)d_in[8];
  float* out = (float*)d_out;
  (void)in_sizes; (void)n_in; (void)out_size; (void)ws_size;

  char* base = (char*)d_ws;
  size_t off = 0;
  auto wsa = [&](size_t bytes) -> void* {
    void* p = base + off;
    off = (off + bytes + 255) & ~(size_t)255;
    return p;
  };
  __bf16* xbf  = (__bf16*)wsa((size_t)B * Nn * Cc * 2);       // ~38.6 MB
  __bf16* wkbf = (__bf16*)wsa((size_t)Cc * Cc * 2);
  __bf16* wvbf = (__bf16*)wsa((size_t)Cc * Cc * 2);
  __bf16* pwbf = (__bf16*)wsa((size_t)Cc * Cc * 2);
  __bf16* qbf  = (__bf16*)wsa((size_t)Hh * PPAD * HD * 2);
  __bf16* Kbf  = (__bf16*)wsa((size_t)B * Hh * NPAD * HD * 2); // ~40.9 MB
  __bf16* Vt   = (__bf16*)wsa((size_t)B * Hh * HD * NPAD * 2); // ~40.9 MB
  float*  bias = (float*)wsa((size_t)Hh * PPAD * NPAD * 4);    // ~4.2 MB
  __bf16* Obf  = (__bf16*)wsa((size_t)B * PPAD * Cc * 2);      // ~20.4 MB

  const int nx = B * Nn * Cc;
  const int nw = Cc * Cc;
  cvt_bf16_kernel<<<(nx + 255) / 256, 256, 0, stream>>>(x, xbf, nx);
  cvt_bf16_kernel<<<(nw + 255) / 256, 256, 0, stream>>>(wk, wkbf, nw);
  cvt_bf16_kernel<<<(nw + 255) / 256, 256, 0, stream>>>(wv, wvbf, nw);
  cvt_bf16_kernel<<<(nw + 255) / 256, 256, 0, stream>>>(proj_w, pwbf, nw);
  prep_q_bias_kernel<<<dim3(PPAD, Hh), 128, 0, stream>>>(
      q_learned, pos_embed, rpe_w, rp_bucket, qbf, bias);
  kv_proj_kernel<<<dim3(NTILES, B), 128, 0, stream>>>(xbf, wkbf, wvbf, Kbf, Vt);
  attn_kernel<<<dim3(PTILES, Hh, B), 128, 0, stream>>>(qbf, Kbf, Vt, bias, Obf);
  out_proj_kernel<<<dim3(PTILES, B), 128, 0, stream>>>(Obf, pwbf, proj_b, out);
}
